// ProteinMPNN_13262859010369
// MI455X (gfx1250) — matrix operand order repaired and checked by reading, not compiled, not run on hardware
//
#include <hip/hip_runtime.h>
#include <math.h>

#define BB 4
#define LL 2048
#define KNN 30
#define HD 128
#define NPE 16
#define NRBF 16
#define MAXREL 32
#define EDGE_IN 167   // 16 + 144 + 7

typedef __attribute__((ext_vector_type(16))) __bf16 v16bf;
typedef __attribute__((ext_vector_type(8)))  __bf16 v8bf;
typedef __attribute__((ext_vector_type(8)))  float  v8f;

// ---------- helpers ----------
__device__ __forceinline__ unsigned short f2bf(float x) {
    union { float f; unsigned u; } v; v.f = x;
    unsigned b = v.u;
    unsigned r = b + 0x7fffu + ((b >> 16) & 1u);   // round-to-nearest-even
    return (unsigned short)(r >> 16);
}
__device__ __forceinline__ float geluf(float x) {
    return 0.5f * x * (1.0f + erff(x * 0.70710678118654752f));
}
__device__ __forceinline__ float signf(float x) {
    return (x > 0.f ? 1.f : 0.f) - (x < 0.f ? 1.f : 0.f);
}

__device__ __forceinline__ v8f wmma_bf16(v16bf a, v16bf b, v8f c) {
    return __builtin_amdgcn_wmma_f32_16x16x32_bf16(false, a, false, b, (short)0, c, false, false);
}

// A-fragment (16x32): lane holds row (lane&15), contiguous 16 K at kc+(lane>>4)*16.
__device__ __forceinline__ v16bf load_afrag(const unsigned short* A, int KD_, int row, int kc, int lane) {
    return *(const v16bf*)&A[row * KD_ + kc + ((lane >> 4) << 4)];
}
// B-fragment (32x16): lane holds column (lane&15), contiguous 16 K at kc+(lane>>4)*16.
// Weights are pre-transposed bf16 [N][K].
__device__ __forceinline__ v16bf load_bfrag(const unsigned short* Wt, int KD_, int col, int kc, int lane) {
    return *(const v16bf*)&Wt[(size_t)col * KD_ + kc + ((lane >> 4) << 4)];
}

template<int KD>
__device__ __forceinline__ void gemm2x2(const unsigned short* A, const unsigned short* Wt,
                                        int nb, int lane, v8f acc[4]) {
    int m = lane & 15;
    for (int kc = 0; kc < KD; kc += 32) {
        v16bf a0 = load_afrag(A, KD, m, kc, lane);
        v16bf a1 = load_afrag(A, KD, 16 + m, kc, lane);
        v16bf b0 = load_bfrag(Wt, KD, nb + m, kc, lane);
        v16bf b1 = load_bfrag(Wt, KD, nb + 16 + m, kc, lane);
        acc[0] = wmma_bf16(a0, b0, acc[0]);
        acc[1] = wmma_bf16(a0, b1, acc[1]);
        acc[2] = wmma_bf16(a1, b0, acc[2]);
        acc[3] = wmma_bf16(a1, b1, acc[3]);
    }
}

// D tile epilogues (C/D layout: row = mrow0 + (lane>>4)*8 + r, col = ncol0 + (lane&15))
__device__ __forceinline__ void store_tile_gelu_bf16(v8f acc, int mrow0, int ncol0, int lane,
                                                     const float* bias, unsigned short* out, int ldo) {
    int cs = lane & 15, ks = lane >> 4;
    int col = ncol0 + cs;
    float bc = bias[col];
#pragma unroll
    for (int r = 0; r < 8; r++) {
        int row = mrow0 + ks * 8 + r;
        out[row * ldo + col] = f2bf(geluf(acc[r] + bc));
    }
}
__device__ __forceinline__ void store_tile_f32(v8f acc, int mrow0, int ncol0, int lane,
                                               const float* bias, float* out, int ldo) {
    int cs = lane & 15, ks = lane >> 4;
    int col = ncol0 + cs;
    float bc = bias[col];
#pragma unroll
    for (int r = 0; r < 8; r++) {
        int row = mrow0 + ks * 8 + r;
        out[row * ldo + col] = acc[r] + bc;
    }
}

// ---------- weight prep: fp32 [K][N] -> bf16 transposed [N][K] ----------
__global__ void wprep_kernel(const float* __restrict__ src, unsigned short* __restrict__ dst,
                             int Kd, int Nd) {
    int t = blockIdx.x * blockDim.x + threadIdx.x;
    if (t >= Kd * Nd) return;
    int n = t / Kd, k = t % Kd;
    dst[t] = f2bf(src[(size_t)k * Nd + n]);
}

__global__ void zero_kernel(float* p, int n) {
    int t = blockIdx.x * blockDim.x + threadIdx.x;
    if (t < n) p[t] = 0.f;
}

// ---------- brute-force masked distance + top-K ----------
__global__ __launch_bounds__(256) void topk_kernel(const float* __restrict__ Ca,
                                                   const float* __restrict__ mask,
                                                   int* __restrict__ Eidx, float* __restrict__ Dnb) {
    __shared__ float dist[LL];
    __shared__ float red[256];
    __shared__ int   redi[256];
    int bi = blockIdx.x; int b = bi / LL;
    int tid = threadIdx.x;
    float xi = Ca[(size_t)bi * 3 + 0], yi = Ca[(size_t)bi * 3 + 1], zi = Ca[(size_t)bi * 3 + 2];
    float mi = mask[bi];
    float lmax = -1e30f;
    for (int j = tid; j < LL; j += 256) {
        size_t jj = (size_t)b * LL + j;
        float dx = xi - Ca[jj * 3 + 0], dy = yi - Ca[jj * 3 + 1], dz = zi - Ca[jj * 3 + 2];
        float d = mi * mask[jj] * sqrtf(dx * dx + dy * dy + dz * dz + 1e-6f);
        dist[j] = d;
        lmax = fmaxf(lmax, d);
    }
    red[tid] = lmax; __syncthreads();
    for (int s = 128; s > 0; s >>= 1) { if (tid < s) red[tid] = fmaxf(red[tid], red[tid + s]); __syncthreads(); }
    float dmax = red[0]; __syncthreads();
    for (int j = tid; j < LL; j += 256) {
        float m2 = mi * mask[(size_t)b * LL + j];
        dist[j] = dist[j] + (1.0f - m2) * dmax;
    }
    __syncthreads();
    for (int k = 0; k < KNN; k++) {
        float best = 1e30f; int bidx = LL;
        for (int j = tid; j < LL; j += 256) {
            float d = dist[j];
            if (d < best || (d == best && j < bidx)) { best = d; bidx = j; }
        }
        red[tid] = best; redi[tid] = bidx; __syncthreads();
        for (int s = 128; s > 0; s >>= 1) {
            if (tid < s) {
                float d2 = red[tid + s]; int i2 = redi[tid + s];
                if (d2 < red[tid] || (d2 == red[tid] && i2 < redi[tid])) { red[tid] = d2; redi[tid] = i2; }
            }
            __syncthreads();
        }
        if (tid == 0) {
            Eidx[(size_t)bi * KNN + k] = redi[0];
            Dnb[(size_t)bi * KNN + k] = red[0];
            dist[redi[0]] = 1e30f;
        }
        __syncthreads();
    }
}

// ---------- per-node orientation frames O (3x3 row-major rows = axes) ----------
__global__ void frames_kernel(const float* __restrict__ Ca, float* __restrict__ O) {
    int t = blockIdx.x * blockDim.x + threadIdx.x;
    if (t >= BB * LL) return;
    int b = t / LL, l = t % LL;
    float o[9] = {0.f, 0.f, 0.f, 0.f, 0.f, 0.f, 0.f, 0.f, 0.f};
    if (l >= 1 && l <= LL - 3) {
        const float* base = Ca + (size_t)(b * LL) * 3;
        float d2[3], d1[3];
        for (int c = 0; c < 3; c++) {
            d2[c] = base[(size_t)l * 3 + c] - base[(size_t)(l - 1) * 3 + c];
            d1[c] = base[(size_t)(l + 1) * 3 + c] - base[(size_t)l * 3 + c];
        }
        auto ugate = [](const float* d, float* u) {
            float n = sqrtf(d[0]*d[0] + d[1]*d[1] + d[2]*d[2]);
            float g = (n > 3.6f && n < 4.0f) ? 1.f : 0.f;
            float nn = fmaxf(n * g, 1e-12f);
            u[0] = d[0] * g / nn; u[1] = d[1] * g / nn; u[2] = d[2] * g / nn;
        };
        float u2[3], u1[3];
        ugate(d2, u2); ugate(d1, u1);
        float o1[3] = {u2[0]-u1[0], u2[1]-u1[1], u2[2]-u1[2]};
        float n = fmaxf(sqrtf(o1[0]*o1[0]+o1[1]*o1[1]+o1[2]*o1[2]), 1e-12f);
        o1[0]/=n; o1[1]/=n; o1[2]/=n;
        float n2[3] = {u2[1]*u1[2]-u2[2]*u1[1], u2[2]*u1[0]-u2[0]*u1[2], u2[0]*u1[1]-u2[1]*u1[0]};
        n = fmaxf(sqrtf(n2[0]*n2[0]+n2[1]*n2[1]+n2[2]*n2[2]), 1e-12f);
        n2[0]/=n; n2[1]/=n; n2[2]/=n;
        float c3[3] = {o1[1]*n2[2]-o1[2]*n2[1], o1[2]*n2[0]-o1[0]*n2[2], o1[0]*n2[1]-o1[1]*n2[0]};
        o[0]=o1[0]; o[1]=o1[1]; o[2]=o1[2];
        o[3]=n2[0]; o[4]=n2[1]; o[5]=n2[2];
        o[6]=c3[0]; o[7]=c3[1]; o[8]=c3[2];
    }
    for (int q = 0; q < 9; q++) O[(size_t)t * 9 + q] = o[q];
}

__device__ __forceinline__ void shiftedCa(const float* Ca, int b, int l, int s,
                                          float& x, float& y, float& z) {
    int ls = l + (s - 1);      // s: 0 -> l-1, 1 -> l, 2 -> l+1
    if (ls < 0 || ls >= LL) { x = y = z = 0.f; return; }
    const float* p = Ca + (size_t)(b * LL + ls) * 3;
    x = p[0]; y = p[1]; z = p[2];
}

// ---------- edge features -> E (LN) -> h_E = E @ We + be ----------
__global__ __launch_bounds__(256) void features_kernel(
    const float* __restrict__ Ca, const float* __restrict__ mask,
    const int* __restrict__ ridx, const int* __restrict__ chl,
    const float* __restrict__ pe_w, const float* __restrict__ pe_b,
    const float* __restrict__ edge_w, const float* __restrict__ edge_g, const float* __restrict__ edge_b,
    const float* __restrict__ We_w, const float* __restrict__ We_b,
    const int* __restrict__ Eidx, const float* __restrict__ Dnb, const float* __restrict__ Ofr,
    float* __restrict__ hE) {
    __shared__ float Eraw[EDGE_IN + 1];
    __shared__ float Evec[HD];
    __shared__ float red[256];
    int bi = blockIdx.x; int b = bi / LL, i = bi % LL;
    int tid = threadIdx.x;
    for (int k = 0; k < KNN; k++) {
        int j = Eidx[(size_t)bi * KNN + k];
        if (tid < EDGE_IN) {
            float val = 0.f;
            if (tid < NPE) {
                int off = ridx[bi] - ridx[b * LL + j];
                int ch = (chl[bi] == chl[b * LL + j]) ? 1 : 0;
                int dc = off + MAXREL; if (dc < 0) dc = 0; if (dc > 2 * MAXREL) dc = 2 * MAXREL;
                int d = dc * ch + (1 - ch) * (2 * MAXREL + 1);
                val = pe_w[d * NPE + tid] + pe_b[tid];
            } else if (tid < NPE + 9 * NRBF) {
                int t = tid - NPE; int q = t >> 4; int m = t & 15;
                float D;
                if (q == 0) D = Dnb[(size_t)bi * KNN + k];
                else {
                    const int qa[9] = {1,0,2,0,0,1,1,2,2};
                    const int qb[9] = {1,0,2,1,2,0,2,0,1};
                    float ax, ay, az, bx, by, bz;
                    shiftedCa(Ca, b, i, qa[q], ax, ay, az);
                    shiftedCa(Ca, b, j, qb[q], bx, by, bz);
                    float dx = ax - bx, dy = ay - by, dz = az - bz;
                    D = sqrtf(dx * dx + dy * dy + dz * dz + 1e-6f);
                }
                float mu = 2.0f + (20.0f / 15.0f) * (float)m;
                float z = (D - mu) * 0.8f;   // /1.25
                val = expf(-z * z);
            } else {
                int t = tid - (NPE + 9 * NRBF);    // 0..6
                float Oi[9], Oj[9];
                for (int q2 = 0; q2 < 9; q2++) {
                    Oi[q2] = Ofr[(size_t)bi * 9 + q2];
                    Oj[q2] = Ofr[(size_t)(b * LL + j) * 9 + q2];
                }
                float dX[3];
                for (int c = 0; c < 3; c++)
                    dX[c] = Ca[(size_t)(b * LL + j) * 3 + c] - Ca[(size_t)bi * 3 + c];
                if (t < 3) {
                    float du[3];
                    for (int r = 0; r < 3; r++)
                        du[r] = Oi[r*3+0]*dX[0] + Oi[r*3+1]*dX[1] + Oi[r*3+2]*dX[2];
                    float n = fmaxf(sqrtf(du[0]*du[0]+du[1]*du[1]+du[2]*du[2]), 1e-12f);
                    val = du[t] / n;
                } else {
                    float R[3][3];
                    for (int a = 0; a < 3; a++)
                        for (int c = 0; c < 3; c++)
                            R[a][c] = Oi[0*3+a]*Oj[0*3+c] + Oi[1*3+a]*Oj[1*3+c] + Oi[2*3+a]*Oj[2*3+c];
                    float Rxx = R[0][0], Ryy = R[1][1], Rzz = R[2][2];
                    float m0 = 0.5f * sqrtf(fabsf(1.f + Rxx - Ryy - Rzz));
                    float m1 = 0.5f * sqrtf(fabsf(1.f - Rxx + Ryy - Rzz));
                    float m2 = 0.5f * sqrtf(fabsf(1.f - Rxx - Ryy + Rzz));
                    float qv[4];
                    qv[0] = signf(R[2][1] - R[1][2]) * m0;
                    qv[1] = signf(R[0][2] - R[2][0]) * m1;
                    qv[2] = signf(R[1][0] - R[0][1]) * m2;
                    qv[3] = sqrtf(fmaxf(0.f, 1.f + Rxx + Ryy + Rzz)) * 0.5f;
                    float n = fmaxf(sqrtf(qv[0]*qv[0]+qv[1]*qv[1]+qv[2]*qv[2]+qv[3]*qv[3]), 1e-12f);
                    val = qv[t - 3] / n;
                }
            }
            Eraw[tid] = val;
        }
        __syncthreads();
        float acc = 0.f;
        if (tid < HD) {
            for (int c = 0; c < EDGE_IN; c++) acc += Eraw[c] * edge_w[c * HD + tid];
        }
        red[tid] = (tid < HD) ? acc : 0.f; __syncthreads();
        for (int s = 128; s > 0; s >>= 1) { if (tid < s) red[tid] += red[tid + s]; __syncthreads(); }
        float mu = red[0] * (1.f / 128.f); __syncthreads();
        float dv = (tid < HD) ? (acc - mu) : 0.f;
        red[tid] = dv * dv; __syncthreads();
        for (int s = 128; s > 0; s >>= 1) { if (tid < s) red[tid] += red[tid + s]; __syncthreads(); }
        float var = red[0] * (1.f / 128.f); __syncthreads();
        if (tid < HD) Evec[tid] = dv * rsqrtf(var + 1e-5f) * edge_g[tid] + edge_b[tid];
        __syncthreads();
        if (tid < HD) {
            float he = We_b[tid];
            for (int c = 0; c < HD; c++) he += Evec[c] * We_w[c * HD + tid];
            hE[((size_t)bi * KNN + k) * HD + tid] = he;
        }
        __syncthreads();
    }
}

// ---------- fused edge MLP (node message: mode 0; edge update: mode 1) ----------
__global__ __launch_bounds__(128) void edge_mlp_kernel(
    const float* __restrict__ hV, float* __restrict__ hE,
    const int* __restrict__ Eidx, const float* __restrict__ mask,
    const unsigned short* __restrict__ W1t, const float* __restrict__ b1,
    const unsigned short* __restrict__ W2t, const float* __restrict__ b2,
    const unsigned short* __restrict__ W3t, const float* __restrict__ b3,
    const float* __restrict__ g, const float* __restrict__ be,
    float* __restrict__ hV_out, int mode) {
    __shared__ unsigned short Abuf[32 * 384];
    __shared__ unsigned short H1[32 * HD];
    __shared__ unsigned short H2[32 * HD];
    __shared__ float Mbuf[32 * HD];
    __shared__ float red[128];
    __shared__ int jidx[32];

    int bi = blockIdx.x; int b = bi / LL;
    int tid = threadIdx.x, w = tid >> 5, lane = tid & 31;
    size_t nodebase = (size_t)bi * HD;
    size_t edgebase = (size_t)bi * KNN * HD;

    if (tid < 32) jidx[tid] = (tid < KNN) ? Eidx[(size_t)bi * KNN + tid] : 0;
    __syncthreads();
    // stage A = [h_V_i | h_E | h_V_j] (rows 30,31 zero)
    for (int idx = tid; idx < 32 * 384; idx += 128) {
        int r = idx / 384, c = idx % 384;
        float v = 0.f;
        if (r < KNN) {
            if (c < 128)      v = hV[nodebase + c];
            else if (c < 256) v = hE[edgebase + (size_t)r * HD + (c - 128)];
            else              v = hV[(size_t)(b * LL + jidx[r]) * HD + (c - 256)];
        }
        Abuf[idx] = f2bf(v);
    }
    __syncthreads();

    v8f z8 = {0.f,0.f,0.f,0.f,0.f,0.f,0.f,0.f};
    {   // GEMM1: 384 -> 128, gelu
        v8f acc[4] = {z8, z8, z8, z8};
        gemm2x2<384>(Abuf, W1t, w * 32, lane, acc);
        store_tile_gelu_bf16(acc[0], 0,  w*32,      lane, b1, H1, HD);
        store_tile_gelu_bf16(acc[1], 0,  w*32 + 16, lane, b1, H1, HD);
        store_tile_gelu_bf16(acc[2], 16, w*32,      lane, b1, H1, HD);
        store_tile_gelu_bf16(acc[3], 16, w*32 + 16, lane, b1, H1, HD);
    }
    __syncthreads();
    {   // GEMM2: 128 -> 128, gelu
        v8f acc[4] = {z8, z8, z8, z8};
        gemm2x2<128>(H1, W2t, w * 32, lane, acc);
        store_tile_gelu_bf16(acc[0], 0,  w*32,      lane, b2, H2, HD);
        store_tile_gelu_bf16(acc[1], 0,  w*32 + 16, lane, b2, H2, HD);
        store_tile_gelu_bf16(acc[2], 16, w*32,      lane, b2, H2, HD);
        store_tile_gelu_bf16(acc[3], 16, w*32 + 16, lane, b2, H2, HD);
    }
    __syncthreads();
    {   // GEMM3: 128 -> 128, + bias, f32
        v8f acc[4] = {z8, z8, z8, z8};
        gemm2x2<128>(H2, W3t, w * 32, lane, acc);
        store_tile_f32(acc[0], 0,  w*32,      lane, b3, Mbuf, HD);
        store_tile_f32(acc[1], 0,  w*32 + 16, lane, b3, Mbuf, HD);
        store_tile_f32(acc[2], 16, w*32,      lane, b3, Mbuf, HD);
        store_tile_f32(acc[3], 16, w*32 + 16, lane, b3, Mbuf, HD);
    }
    __syncthreads();

    if (mode == 0) {
        // masked mean over K, residual, LN over 128 features
        float mi = mask[bi];
        float s = 0.f;
        for (int r = 0; r < KNN; r++) {
            float ma = mi * mask[b * LL + jidx[r]];
            s += ma * Mbuf[r * HD + tid];
        }
        s *= (1.0f / 30.0f);
        float x = hV[nodebase + tid] + s;
        red[tid] = x; __syncthreads();
        for (int st = 64; st > 0; st >>= 1) { if (tid < st) red[tid] += red[tid + st]; __syncthreads(); }
        float mu = red[0] * (1.f / 128.f); __syncthreads();
        float d = x - mu;
        red[tid] = d * d; __syncthreads();
        for (int st = 64; st > 0; st >>= 1) { if (tid < st) red[tid] += red[tid + st]; __syncthreads(); }
        float var = red[0] * (1.f / 128.f);
        hV_out[nodebase + tid] = d * rsqrtf(var + 1e-5f) * g[tid] + be[tid];
    } else {
        // per-edge residual LN (wave per row, wave32 shuffle reductions)
        for (int r = w; r < KNN; r += 4) {
            float xv[4]; float s = 0.f;
#pragma unroll
            for (int c4 = 0; c4 < 4; c4++) {
                int c = lane * 4 + c4;
                float x = hE[edgebase + (size_t)r * HD + c] + Mbuf[r * HD + c];
                xv[c4] = x; s += x;
            }
            for (int off = 16; off > 0; off >>= 1) s += __shfl_xor(s, off, 32);
            float mu = s * (1.f / 128.f);
            float vs = 0.f;
#pragma unroll
            for (int c4 = 0; c4 < 4; c4++) { float d = xv[c4] - mu; vs += d * d; }
            for (int off = 16; off > 0; off >>= 1) vs += __shfl_xor(vs, off, 32);
            float rs = rsqrtf(vs * (1.f / 128.f) + 1e-5f);
#pragma unroll
            for (int c4 = 0; c4 < 4; c4++) {
                int c = lane * 4 + c4;
                hE[edgebase + (size_t)r * HD + c] = (xv[c4] - mu) * rs * g[c] + be[c];
            }
        }
    }
}

// ---------- fused FFN (128 -> 512 gelu -> 128) + residual LN + mask ----------
__global__ __launch_bounds__(128) void ffn_kernel(
    const float* __restrict__ hVmid, const float* __restrict__ mask,
    const unsigned short* __restrict__ Wint, const float* __restrict__ bin,
    const unsigned short* __restrict__ Woutt, const float* __restrict__ bout,
    const float* __restrict__ g, const float* __restrict__ be,
    float* __restrict__ hV_out) {
    __shared__ unsigned short Abuf[32 * HD];
    __shared__ unsigned short H1[32 * 512];
    __shared__ float Obuf[32 * HD];
    int n0 = blockIdx.x * 32;
    int tid = threadIdx.x, w = tid >> 5, lane = tid & 31;

    for (int idx = tid; idx < 32 * HD; idx += 128)
        Abuf[idx] = f2bf(hVmid[(size_t)n0 * HD + idx]);
    __syncthreads();

    v8f z8 = {0.f,0.f,0.f,0.f,0.f,0.f,0.f,0.f};
    // GEMM1: 128 -> 512 (wave w owns cols [w*128, w*128+128) in 4 passes of 32)
    for (int p = 0; p < 4; p++) {
        v8f acc[4] = {z8, z8, z8, z8};
        int nb = w * 128 + p * 32;
        gemm2x2<128>(Abuf, Wint, nb, lane, acc);
        store_tile_gelu_bf16(acc[0], 0,  nb,      lane, bin, H1, 512);
        store_tile_gelu_bf16(acc[1], 0,  nb + 16, lane, bin, H1, 512);
        store_tile_gelu_bf16(acc[2], 16, nb,      lane, bin, H1, 512);
        store_tile_gelu_bf16(acc[3], 16, nb + 16, lane, bin, H1, 512);
    }
    __syncthreads();
    {   // GEMM2: 512 -> 128
        v8f acc[4] = {z8, z8, z8, z8};
        gemm2x2<512>(H1, Woutt, w * 32, lane, acc);
        store_tile_f32(acc[0], 0,  w*32,      lane, bout, Obuf, HD);
        store_tile_f32(acc[1], 0,  w*32 + 16, lane, bout, Obuf, HD);
        store_tile_f32(acc[2], 16, w*32,      lane, bout, Obuf, HD);
        store_tile_f32(acc[3], 16, w*32 + 16, lane, bout, Obuf, HD);
    }
    __syncthreads();
    // residual + LN + mask, wave per row
    for (int r = w; r < 32; r += 4) {
        size_t node = (size_t)(n0 + r);
        float mk = mask[node];
        float xv[4]; float s = 0.f;
#pragma unroll
        for (int c4 = 0; c4 < 4; c4++) {
            int c = lane * 4 + c4;
            float x = hVmid[node * HD + c] + Obuf[r * HD + c];
            xv[c4] = x; s += x;
        }
        for (int off = 16; off > 0; off >>= 1) s += __shfl_xor(s, off, 32);
        float mu = s * (1.f / 128.f);
        float vs = 0.f;
#pragma unroll
        for (int c4 = 0; c4 < 4; c4++) { float d = xv[c4] - mu; vs += d * d; }
        for (int off = 16; off > 0; off >>= 1) vs += __shfl_xor(vs, off, 32);
        float rs = rsqrtf(vs * (1.f / 128.f) + 1e-5f);
#pragma unroll
        for (int c4 = 0; c4 < 4; c4++) {
            int c = lane * 4 + c4;
            hV_out[node * HD + c] = mk * ((xv[c4] - mu) * rs * g[c] + be[c]);
        }
    }
}

// ---------- host orchestration ----------
extern "C" void kernel_launch(void* const* d_in, const int* in_sizes, int n_in,
                              void* d_out, int out_size, void* d_ws, size_t ws_size,
                              hipStream_t stream) {
    (void)in_sizes; (void)n_in; (void)out_size; (void)ws_size;
    const float* Ca     = (const float*)d_in[0];
    const float* mask   = (const float*)d_in[1];
    const int*   ridx   = (const int*)d_in[2];
    const int*   chl    = (const int*)d_in[3];
    const float* pe_w   = (const float*)d_in[4];
    const float* pe_b   = (const float*)d_in[5];
    const float* edge_w = (const float*)d_in[6];
    const float* edge_g = (const float*)d_in[7];
    const float* edge_b = (const float*)d_in[8];
    const float* We_w   = (const float*)d_in[9];
    const float* We_b   = (const float*)d_in[10];

    char* ws = (char*)d_ws;
    size_t off = 0;
    auto carve = [&](size_t bytes) -> void* {
        void* p = ws + off;
        off = (off + bytes + 255) & ~(size_t)255;
        return p;
    };
    const size_t BLK = (size_t)BB * LL * KNN;
    int*            Eidx  = (int*)carve(BLK * 4);
    float*          Dnb   = (float*)carve(BLK * 4);
    float*          Ofr   = (float*)carve((size_t)BB * LL * 9 * 4);
    float*          hE    = (float*)carve(BLK * HD * 4);
    float*          hV    = (float*)carve((size_t)BB * LL * HD * 4);
    float*          hVmid = (float*)carve((size_t)BB * LL * HD * 4);
    unsigned short* Wbf   = (unsigned short*)carve((size_t)3 * 294912 * 2);

    // per-layer bf16 weight offsets within 294912-element stride
    // W1t:0  W2t:49152  W3t:65536  Wint:81920  Woutt:147456  W11t:212992  W12t:262144  W13t:278528
    for (int l = 0; l < 3; l++) {
        int pb = 11 + 22 * l;
        unsigned short* wb = Wbf + (size_t)l * 294912;
        auto prep = [&](int pi, unsigned short* dst, int Kd, int Nd) {
            int n = Kd * Nd;
            wprep_kernel<<<(n + 255) / 256, 256, 0, stream>>>((const float*)d_in[pi], dst, Kd, Nd);
        };
        prep(pb + 0,  wb + 0,      384, 128);   // W1
        prep(pb + 2,  wb + 49152,  128, 128);   // W2
        prep(pb + 4,  wb + 65536,  128, 128);   // W3
        prep(pb + 8,  wb + 81920,  128, 512);   // Win  -> [512][128]
        prep(pb + 10, wb + 147456, 512, 128);   // Wout -> [128][512]
        prep(pb + 14, wb + 212992, 384, 128);   // W11
        prep(pb + 16, wb + 262144, 128, 128);   // W12
        prep(pb + 18, wb + 278528, 128, 128);   // W13
    }

    topk_kernel<<<BB * LL, 256, 0, stream>>>(Ca, mask, Eidx, Dnb);
    frames_kernel<<<(BB * LL + 255) / 256, 256, 0, stream>>>(Ca, Ofr);
    features_kernel<<<BB * LL, 256, 0, stream>>>(Ca, mask, ridx, chl, pe_w, pe_b,
                                                 edge_w, edge_g, edge_b, We_w, We_b,
                                                 Eidx, Dnb, Ofr, hE);
    zero_kernel<<<(BB * LL * HD + 255) / 256, 256, 0, stream>>>(hV, BB * LL * HD);

    for (int l = 0; l < 3; l++) {
        int pb = 11 + 22 * l;
        unsigned short* wb = Wbf + (size_t)l * 294912;
        // node message + residual LN (g1, be1) -> hVmid
        edge_mlp_kernel<<<BB * LL, 128, 0, stream>>>(
            hV, hE, Eidx, mask,
            wb + 0,      (const float*)d_in[pb + 1],
            wb + 49152,  (const float*)d_in[pb + 3],
            wb + 65536,  (const float*)d_in[pb + 5],
            (const float*)d_in[pb + 6], (const float*)d_in[pb + 7],
            hVmid, 0);
        // FFN + residual LN (g2, be2) + mask -> hV
        ffn_kernel<<<(BB * LL) / 32, 128, 0, stream>>>(
            hVmid, mask,
            wb + 81920,  (const float*)d_in[pb + 9],
            wb + 147456, (const float*)d_in[pb + 11],
            (const float*)d_in[pb + 12], (const float*)d_in[pb + 13],
            hV);
        // edge update + residual LN (g3, be3), hE in place
        edge_mlp_kernel<<<BB * LL, 128, 0, stream>>>(
            hV, hE, Eidx, mask,
            wb + 212992, (const float*)d_in[pb + 15],
            wb + 262144, (const float*)d_in[pb + 17],
            wb + 278528, (const float*)d_in[pb + 19],
            (const float*)d_in[pb + 20], (const float*)d_in[pb + 21],
            nullptr, 1);
    }

    hipMemcpyAsync(d_out, hV, (size_t)BB * LL * HD * 4, hipMemcpyDeviceToDevice, stream);
}
